// GCN_2001454760610
// MI455X (gfx1250) — matrix operand order repaired and checked by reading, not compiled, not run on hardware
//
#include <hip/hip_runtime.h>
#include <hip/hip_bf16.h>

typedef __attribute__((ext_vector_type(16))) __bf16 v16bf;
typedef __attribute__((ext_vector_type(8)))  float  v8f;
typedef __attribute__((ext_vector_type(4)))  unsigned u32x4;
typedef __attribute__((ext_vector_type(8)))  int      i32x8;
typedef __attribute__((ext_vector_type(4)))  int      i32x4;

#define DIM      512
#define NGRAPHS  128
#define NLAYERS  4
#define BN_EPS   1e-5f

// GEMM tiling
#define BM 64
#define TN 256
#define BK 32
#define LDA 17   // uints (bf16 pairs) per A-tile row, padded (TDM pad: +1 DWORD / 16 DWORDs)
#define LDB 17   // uints per B-tile row (transposed), padded

// ---------- helpers ----------
__device__ __forceinline__ unsigned pack2_bf16(float a, float b) {
  unsigned ua = __builtin_bit_cast(unsigned, a);
  unsigned ub = __builtin_bit_cast(unsigned, b);
  ua += 0x7FFFu + ((ua >> 16) & 1u);            // RNE to bf16
  ub += 0x7FFFu + ((ub >> 16) & 1u);
  return (ua >> 16) | (ub & 0xFFFF0000u);
}
__device__ __forceinline__ unsigned short f2bf(float x) {
  unsigned u = __builtin_bit_cast(unsigned, x);
  u += 0x7FFFu + ((u >> 16) & 1u);
  return (unsigned short)(u >> 16);
}
__device__ __forceinline__ unsigned lds_off(const void* p) {
  return (unsigned)(uintptr_t)p;                 // low 32 bits of generic ptr = LDS offset
}

// TDM 2D tile load, data_size=4B, LDS padding: 1 DWORD after every 16 DWORDs.
// tensor dims give OOB-zeroing; strides in DWORD units.
__device__ __forceinline__ void tdm_load_2d(unsigned ldsAddr, unsigned long long gaddr,
                                            unsigned tensor_d0, unsigned tensor_d1,
                                            unsigned tile_d0, unsigned tile_d1,
                                            unsigned long long stride0) {
  u32x4 g0;
  g0.x = 1u;                                               // count=1, user mode
  g0.y = ldsAddr;                                          // lds_addr [63:32]
  g0.z = (unsigned)gaddr;                                  // global_addr [95:64]
  g0.w = ((unsigned)(gaddr >> 32) & 0x1FFFFFFu) | (2u << 30); // addr[56:32], type=2
  i32x8 g1;
  g1[0] = (int)((2u << 16) | (1u << 20) | (3u << 22));     // data_size=4B, pad_en, every 16 DW, pad 1 DW
  g1[1] = (int)((tensor_d0 & 0xFFFFu) << 16);              // tensor_dim0 lo16 at [63:48]
  g1[2] = (int)(((tensor_d0 >> 16) & 0xFFFFu) | ((tensor_d1 & 0xFFFFu) << 16));
  g1[3] = (int)(((tensor_d1 >> 16) & 0xFFFFu) | ((tile_d0 & 0xFFFFu) << 16));
  g1[4] = (int)(tile_d1 & 0xFFFFu);                        // tile_dim1 (tile_dim2=0)
  g1[5] = (int)(unsigned)stride0;                          // tensor_dim0_stride lo32
  g1[6] = (int)((unsigned)(stride0 >> 32) & 0xFFFFu);      // stride hi16 (dim1_stride=0)
  g1[7] = 0;
  i32x4 z4 = {0, 0, 0, 0};
#if __clang_major__ >= 23
  i32x8 z8 = {0, 0, 0, 0, 0, 0, 0, 0};
  __builtin_amdgcn_tensor_load_to_lds(g0, g1, z4, z4, z8, 0);
#else
  __builtin_amdgcn_tensor_load_to_lds(g0, g1, z4, z4, 0);
#endif
}

// ---------- weight pre-pack: f32 [512][512] -> bf16-pair tiles [nb][kk][n][p] ----------
// 8 matrices (l*2+w), each 2*16*256*16 = 131072 uints (512 KB).
__global__ __launch_bounds__(256) void pack_weights_kernel(
    const float* __restrict__ W1, const float* __restrict__ W2,
    unsigned* __restrict__ Wp) {
  int i   = blockIdx.x * 256 + threadIdx.x;   // 0 .. 8*131072-1 (exact grid)
  int mat = i >> 17;
  int rem = i & 131071;
  int nb  = rem >> 16;
  int kk  = (rem >> 12) & 15;
  int n   = (rem >> 4) & 255;
  int p   = rem & 15;
  int l   = mat >> 1;
  const float* src = ((mat & 1) ? W2 : W1) + (size_t)l * DIM * DIM;
  int k   = kk * BK + 2 * p;
  int col = nb * TN + n;
  float x0 = src[(size_t)k * DIM + col];
  float x1 = src[(size_t)(k + 1) * DIM + col];
  Wp[i] = pack2_bf16(x0, x1);
}

// ---------- edge aggregation: h[dst] += z[src]  (h pre-initialized to z) ----------
__global__ __launch_bounds__(128) void edge_agg_kernel(
    const float* __restrict__ z, float* __restrict__ h,
    const int* __restrict__ ei, int E) {
  int e  = blockIdx.x;
  int d0 = threadIdx.x * 4;
  int s  = ei[e];
  int d  = ei[E + e];
  const float4 v = *reinterpret_cast<const float4*>(z + (size_t)s * DIM + d0);
  float* hp = h + (size_t)d * DIM + d0;
  atomicAdd(hp + 0, v.x);
  atomicAdd(hp + 1, v.y);
  atomicAdd(hp + 2, v.z);
  atomicAdd(hp + 3, v.w);
}

// ---------- WMMA compute core (shared by both GEMMs) ----------
union FU { unsigned u[8]; v16bf v; };

__device__ __forceinline__ void wmma_step(const unsigned* As, const unsigned* Bs,
                                          int waveM, int waveN, int half, int l15,
                                          v8f acc[2][4]) {
  FU fa[2], fb[4];
#pragma unroll
  for (int mi = 0; mi < 2; ++mi) {
    int m_l = waveM * 32 + mi * 16 + l15;
#pragma unroll
    for (int r = 0; r < 8; ++r) {
      int p = half * 8 + r;                             // A layout: K = 2p
      fa[mi].u[r] = As[m_l * LDA + p];
    }
  }
#pragma unroll
  for (int ni = 0; ni < 4; ++ni) {
    int n_l = waveN * 64 + ni * 16 + l15;
#pragma unroll
    for (int r = 0; r < 8; ++r) {
      int p = half * 8 + r;                             // B layout: K = 2p
      fb[ni].u[r] = Bs[n_l * LDB + p];
    }
  }
#pragma unroll
  for (int mi = 0; mi < 2; ++mi)
#pragma unroll
    for (int ni = 0; ni < 4; ++ni)
      acc[mi][ni] = __builtin_amdgcn_wmma_f32_16x16x32_bf16(
          false, fa[mi].v, false, fb[ni].v, (short)0, acc[mi][ni], false, false);
}

// ---------- GEMM1: t(bf16) = relu(h(f32) @ W1 + b1); B tile via TDM ----------
__global__ __launch_bounds__(256) void gemm1_kernel(
    const float* __restrict__ A, const unsigned* __restrict__ Bp,
    const float* __restrict__ bias, unsigned short* __restrict__ Cbf, int M) {
  __shared__ unsigned As[BM * LDA];
  __shared__ unsigned Bs[TN * LDB];
  const int tid   = threadIdx.x;
  const int lane  = tid & 31;
  const int wid   = tid >> 5;
  const int waveM = wid >> 2, waveN = wid & 3;
  const int half  = lane >> 4, l15 = lane & 15;
  const int m0    = blockIdx.x * BM;
  const unsigned bsOff = lds_off(&Bs[0]);

  v8f acc[2][4];
#pragma unroll
  for (int mi = 0; mi < 2; ++mi)
#pragma unroll
    for (int ni = 0; ni < 4; ++ni)
#pragma unroll
      for (int r = 0; r < 8; ++r) acc[mi][ni][r] = 0.0f;

  for (int kk = 0; kk < DIM / BK; ++kk) {
    if (tid < 32) {   // wave 0: DMA prepacked 16KB B tile (pad -> stride 17)
      unsigned long long gb =
          (unsigned long long)(uintptr_t)(Bp + ((size_t)blockIdx.y << 16) + ((size_t)kk << 12));
      tdm_load_2d(bsOff, gb, 4096u, 1u, 4096u, 1u, 4096ull);
    }
    // all waves: stage + convert A tile (64 x 32 f32 -> bf16 pairs)
#pragma unroll
    for (int j = 0; j < 4; ++j) {
      int idx = tid + j * 256;           // 0..1023
      int m = idx >> 4, p = idx & 15;
      int row = m0 + m;
      float x0 = 0.0f, x1 = 0.0f;
      if (row < M) {
        const float* g = A + (size_t)row * DIM + kk * BK + 2 * p;
        x0 = g[0]; x1 = g[1];
      }
      As[m * LDA + p] = pack2_bf16(x0, x1);
    }
    __builtin_amdgcn_s_wait_tensorcnt(0);   // non-wave0: counter already 0
    __syncthreads();
    wmma_step(As, Bs, waveM, waveN, half, l15, acc);
    __syncthreads();
  }

  const int n0 = blockIdx.y * TN;
#pragma unroll
  for (int mi = 0; mi < 2; ++mi)
#pragma unroll
    for (int ni = 0; ni < 4; ++ni) {
      int n = n0 + waveN * 64 + ni * 16 + l15;
      float bv = bias[n];
#pragma unroll
      for (int r = 0; r < 8; ++r) {
        int m = m0 + waveM * 32 + mi * 16 + half * 8 + r;
        if (m < M) {
          float v = acc[mi][ni][r] + bv;
          Cbf[(size_t)m * DIM + n] = f2bf(v > 0.0f ? v : 0.0f);
        }
      }
    }
}

// ---------- GEMM2: h(f32) = relu(t(bf16) @ W2 + b2); A+B via double-buffered TDM ----------
__global__ __launch_bounds__(256) void gemm2_kernel(
    const unsigned short* __restrict__ Abf, const unsigned* __restrict__ Bp,
    const float* __restrict__ bias, float* __restrict__ C, int M) {
  __shared__ unsigned As[2][BM * LDA];
  __shared__ unsigned Bs[2][TN * LDB];
  const int tid   = threadIdx.x;
  const int lane  = tid & 31;
  const int wid   = tid >> 5;
  const int waveM = wid >> 2, waveN = wid & 3;
  const int half  = lane >> 4, l15 = lane & 15;
  const int m0    = blockIdx.x * BM;
  const unsigned asOff[2] = {lds_off(&As[0][0]), lds_off(&As[1][0])};
  const unsigned bsOff[2] = {lds_off(&Bs[0][0]), lds_off(&Bs[1][0])};

  v8f acc[2][4];
#pragma unroll
  for (int mi = 0; mi < 2; ++mi)
#pragma unroll
    for (int ni = 0; ni < 4; ++ni)
#pragma unroll
      for (int r = 0; r < 8; ++r) acc[mi][ni][r] = 0.0f;

  auto issue = [&](int kk, int buf) {
    // A tile: 64 rows x 16 DWORDs (32 bf16) from bf16 row-major t; OOB rows -> 0
    unsigned long long ga =
        (unsigned long long)(uintptr_t)(Abf + (size_t)m0 * DIM + kk * BK);
    tdm_load_2d(asOff[buf], ga, 16u, (unsigned)(M - m0), 16u, (unsigned)BM, 256ull);
    // B tile: contiguous prepacked 16KB
    unsigned long long gb =
        (unsigned long long)(uintptr_t)(Bp + ((size_t)blockIdx.y << 16) + ((size_t)kk << 12));
    tdm_load_2d(bsOff[buf], gb, 4096u, 1u, 4096u, 1u, 4096ull);
  };

  if (tid < 32) issue(0, 0);
  for (int kk = 0; kk < DIM / BK; ++kk) {
    int cur = kk & 1;
    if (kk < DIM / BK - 1) {
      if (tid < 32) issue(kk + 1, cur ^ 1);         // prefetch next pair
      __builtin_amdgcn_s_wait_tensorcnt(2);         // current pair complete
    } else {
      __builtin_amdgcn_s_wait_tensorcnt(0);
    }
    __syncthreads();
    wmma_step(As[cur], Bs[cur], waveM, waveN, half, l15, acc);
    __syncthreads();
  }

  const int n0 = blockIdx.y * TN;
#pragma unroll
  for (int mi = 0; mi < 2; ++mi)
#pragma unroll
    for (int ni = 0; ni < 4; ++ni) {
      int n = n0 + waveN * 64 + ni * 16 + l15;
      float bv = bias[n];
#pragma unroll
      for (int r = 0; r < 8; ++r) {
        int m = m0 + waveM * 32 + mi * 16 + half * 8 + r;
        if (m < M) {
          float v = acc[mi][ni][r] + bv;
          C[(size_t)m * DIM + n] = v > 0.0f ? v : 0.0f;
        }
      }
    }
}

// ---------- BN column stats ----------
__global__ __launch_bounds__(256) void bn_stats_kernel(
    const float* __restrict__ h, float* __restrict__ sums,
    float* __restrict__ sumsq, int N) {
  int r0 = blockIdx.x * 128;
  int r1 = r0 + 128; if (r1 > N) r1 = N;
  int d  = threadIdx.x;
  float s0 = 0.f, q0 = 0.f, s1 = 0.f, q1 = 0.f;
  for (int r = r0; r < r1; ++r) {
    float a = h[(size_t)r * DIM + d];
    float b = h[(size_t)r * DIM + d + 256];
    s0 += a; q0 += a * a; s1 += b; q1 += b * b;
  }
  atomicAdd(&sums[d], s0);        atomicAdd(&sumsq[d], q0);
  atomicAdd(&sums[d + 256], s1);  atomicAdd(&sumsq[d + 256], q1);
}

__global__ void bn_prep_kernel(const float* __restrict__ sums,
                               const float* __restrict__ sumsq,
                               const float* __restrict__ gamma,
                               const float* __restrict__ beta,
                               float* __restrict__ scale,
                               float* __restrict__ shift, int N) {
  int d = blockIdx.x * blockDim.x + threadIdx.x;
  if (d < DIM) {
    float inv_n = 1.0f / (float)N;
    float mu  = sums[d] * inv_n;
    float var = sumsq[d] * inv_n - mu * mu;
    float inv = rsqrtf(var + BN_EPS);
    float sc  = gamma[d] * inv;
    scale[d] = sc;
    shift[d] = beta[d] - mu * sc;
  }
}

// ---------- normalize + per-graph pool accumulation ----------
__global__ __launch_bounds__(256) void norm_pool_kernel(
    const float* __restrict__ h, const float* __restrict__ scale,
    const float* __restrict__ shift, float* __restrict__ z,
    float* __restrict__ pool, const int* __restrict__ batch) {
  int n = blockIdx.x;
  int g = batch[n];
#pragma unroll
  for (int j = 0; j < 2; ++j) {
    int d = threadIdx.x + j * 256;
    float v = h[(size_t)n * DIM + d] * scale[d] + shift[d];
    z[(size_t)n * DIM + d] = v;
    atomicAdd(&pool[g * DIM + d], v);
  }
}

__global__ void finalize_kernel(const float* __restrict__ pool,
                                const float* __restrict__ counts,
                                float* __restrict__ out, int layer) {
  int i = blockIdx.x * 256 + threadIdx.x;
  if (i < NGRAPHS * DIM) {
    int g = i >> 9, d = i & 511;
    float c = counts[g]; if (c < 1.0f) c = 1.0f;
    out[(size_t)g * (NLAYERS * DIM) + layer * DIM + d] = pool[i] / c;
  }
}

__global__ void zero_f_kernel(float* p, int n) {
  int i = blockIdx.x * 256 + threadIdx.x;
  if (i < n) p[i] = 0.0f;
}

__global__ void counts_kernel(const int* __restrict__ batch,
                              float* __restrict__ counts, int N) {
  int n = blockIdx.x * 256 + threadIdx.x;
  if (n < N) atomicAdd(&counts[batch[n]], 1.0f);
}

// ---------- launcher ----------
extern "C" void kernel_launch(void* const* d_in, const int* in_sizes, int n_in,
                              void* d_out, int out_size, void* d_ws, size_t ws_size,
                              hipStream_t stream) {
  const float* x     = (const float*)d_in[0];
  const float* W1    = (const float*)d_in[1];
  const float* b1    = (const float*)d_in[2];
  const float* W2    = (const float*)d_in[3];
  const float* b2    = (const float*)d_in[4];
  const float* gamma = (const float*)d_in[5];
  const float* beta  = (const float*)d_in[6];
  const int*   ei    = (const int*)d_in[7];
  const int*   batch = (const int*)d_in[8];

  const int N = in_sizes[0] / DIM;
  const int E = in_sizes[7] / 2;
  float* out = (float*)d_out;

  const size_t nd = (size_t)N * DIM;
  auto alignup = [](size_t v) { return (v + 255) & ~(size_t)255; };
  char* ws = (char*)d_ws;
  size_t off = 0;
  float*          zbuf = (float*)(ws + off);          off += alignup(nd * 4);
  float*          hbuf = (float*)(ws + off);          off += alignup(nd * 4);
  unsigned short* tbuf = (unsigned short*)(ws + off); off += alignup(nd * 2);
  unsigned*       Wp   = (unsigned*)(ws + off);       off += alignup((size_t)8 * 131072 * 4);
  float* sums  = (float*)(ws + off);           // 512
  float* sumsq = sums + DIM;                   // 512
  float* pool  = sumsq + DIM;                  // 128*512
  off += alignup((size_t)(2 * DIM + NGRAPHS * DIM) * 4);
  float* scale = (float*)(ws + off);
  float* shift = scale + DIM;
  off += alignup((size_t)(2 * DIM) * 4);
  float* counts = (float*)(ws + off);
  (void)ws_size; (void)n_in; (void)out_size;

  // one-time per launch: weight pre-pack + per-graph counts (deterministic)
  pack_weights_kernel<<<(8 * 131072) / 256, 256, 0, stream>>>(W1, W2, Wp);
  zero_f_kernel<<<1, 256, 0, stream>>>(counts, NGRAPHS);
  counts_kernel<<<(N + 255) / 256, 256, 0, stream>>>(batch, counts, N);

  dim3 gemm_grid((N + BM - 1) / BM, DIM / TN);
  const float* zin = x;
  for (int l = 0; l < NLAYERS; ++l) {
    // h = z ; h[dst] += z[src]  (GIN agg, eps=0)
    hipMemcpyAsync(hbuf, zin, nd * 4, hipMemcpyDeviceToDevice, stream);
    edge_agg_kernel<<<E, 128, 0, stream>>>(zin, hbuf, ei, E);
    // t(bf16) = relu(h @ W1 + b1) ; h = relu(t @ W2 + b2)
    gemm1_kernel<<<gemm_grid, 256, 0, stream>>>(
        hbuf, Wp + (size_t)(l * 2 + 0) * 131072, b1 + l * DIM, tbuf, N);
    gemm2_kernel<<<gemm_grid, 256, 0, stream>>>(
        tbuf, Wp + (size_t)(l * 2 + 1) * 131072, b2 + l * DIM, hbuf, N);
    // BN stats + fold, then normalize + pool
    zero_f_kernel<<<(2 * DIM + NGRAPHS * DIM + 255) / 256, 256, 0, stream>>>(
        sums, 2 * DIM + NGRAPHS * DIM);
    bn_stats_kernel<<<(N + 127) / 128, 256, 0, stream>>>(hbuf, sums, sumsq, N);
    bn_prep_kernel<<<2, 256, 0, stream>>>(sums, sumsq, gamma, beta, scale, shift, N);
    norm_pool_kernel<<<N, 256, 0, stream>>>(hbuf, scale, shift, zbuf, pool, batch);
    finalize_kernel<<<(NGRAPHS * DIM + 255) / 256, 256, 0, stream>>>(pool, counts, out, l);
    zin = zbuf;
  }
}
